// ParticleNet_35287451304394
// MI455X (gfx1250) — matrix operand order repaired and checked by reading, not compiled, not run on hardware
//
#include <hip/hip_runtime.h>
#include <hip/hip_bf16.h>
#include <math.h>

// ---------------------------------------------------------------------------
// ParticleNet forward for MI455X (gfx1250, wave32, WMMA).
// Channel contractions -> v_wmma_f32_16x16x32_f16 with A = weight tile
// (M = out-channels) and B = activation tile (N = data columns), so the D
// layout gives each lane 8 *consecutive* output channels of one column ->
// vectorized float4 epilogue stores. Weights staged once per workgroup into
// LDS as f16; workgroups iterate 128-column chunks with global_prefetch.
// BatchNorm batch statistics -> coalesced linear sweeps + LDS tree reduction.
// ---------------------------------------------------------------------------

typedef _Float16 half_t;
typedef __attribute__((ext_vector_type(16))) _Float16 v16h;
typedef __attribute__((ext_vector_type(8)))  float    v8f;

static constexpr int   B_    = 64;
static constexpr int   P_    = 128;
static constexpr int   NPAIR = 8256;                       // P*(P+1)/2
static constexpr long long NPB = (long long)B_ * NPAIR;    // 528384
static constexpr int   NC    = B_ * P_;                    // 8192 points
static constexpr long long NE = (long long)NC * 127;       // 1040384 edges

// ---- workspace layout (float offsets) -------------------------------------
static constexpr size_t OFF_STATS  = 0;                         // 16 slots x 512
static constexpr size_t OFF_FTS0   = 8192;                      // [NC][16]
static constexpr size_t OFF_FTS1   = OFF_FTS0   + (size_t)NC*16;
static constexpr size_t OFF_FTS2   = OFF_FTS1   + (size_t)NC*32;
static constexpr size_t OFF_FUSIN  = OFF_FTS2   + (size_t)NC*64;
static constexpr size_t OFF_FUSRAW = OFF_FUSIN  + (size_t)NC*96;
static constexpr size_t OFF_POOLB  = OFF_FUSRAW + (size_t)NC*128;
static constexpr size_t OFF_FC1    = OFF_POOLB  + (size_t)B_*128;
static constexpr size_t OFF_SCORES = OFF_FC1    + (size_t)B_*256;
static constexpr size_t OFF_SCRAW  = OFF_SCORES + (size_t)NE;
static constexpr size_t OFF_PW     = OFF_SCRAW  + (size_t)NC*64;
static constexpr size_t OFF_H1     = OFF_PW     + (size_t)NPB*2;
static constexpr size_t OFF_H2     = OFF_H1     + (size_t)NPB*32;
static constexpr size_t OFF_BUFA   = OFF_H2     + (size_t)NPB*64;
static constexpr size_t OFF_BUFB   = OFF_BUFA   + (size_t)NE*64;

// ---- WMMA fragment K-maps (wave32, 16-bit A 16x32 / B 32x16 per ISA) ------
__device__ __forceinline__ int kmapA(int lane, int h) {
  int v = h >> 1, pos = h & 1;
  int base = (v < 4) ? (2 * v) : (16 + 2 * (v - 4));
  return base + ((lane & 16) ? 8 : 0) + pos;
}
__device__ __forceinline__ int kmapB(int lane, int h) {
  return ((h & 8) ? 16 : 0) + (h & 7) + ((lane & 16) ? 8 : 0);
}

// ===========================================================================
// Generic GEMM:  Y[col][COUT] = W(COUT x CIN) * act(X[col][CIN])  (+bias)
// act: 0 = identity, 1 = BN affine (stats slot) + ReLU, 2 = ReLU only.
// CINP = CIN padded to multiple of 32.  A = weight tile, B = activation tile.
// ===========================================================================
template <int CIN, int CINP, int COUT>
__global__ __launch_bounds__(256)
void gemm_wmma_kernel(const float* __restrict__ X, float* __restrict__ Y,
                      const float* __restrict__ W, const float* __restrict__ bias,
                      const float* __restrict__ statsBase, int slot, int actMode,
                      long long Ncols, int chunks) {
  constexpr int KSTEPS = CINP / 32;
  constexpr int NT     = COUT / 16;                 // tiles of output channels
  extern __shared__ half_t lds[];
  half_t* Ws = lds;                       // [COUT][CINP]
  half_t* Xs = lds + COUT * CINP;         // [128][CINP]
  const int tid  = threadIdx.x;
  const int lane = tid & 31;
  const int wave = tid >> 5;
  const int s = slot < 0 ? 0 : slot;
  const float* sc = statsBase + (size_t)s * 512 + 256;
  const float* sh = statsBase + (size_t)s * 512 + 384;

  // stage weights (f16, zero-padded) once
  for (int idx = tid; idx < COUT * CINP; idx += 256) {
    int n = idx / CINP, k = idx % CINP;
    Ws[idx] = (k < CIN) ? (half_t)W[(size_t)n * CIN + k] : (half_t)0.f;
  }

  const long long chunk0 = (long long)blockIdx.x * chunks;
  const int nlane = lane & 15;
  const int colLocal = wave * 16 + nlane;

  for (int cc = 0; cc < chunks; ++cc) {
    const long long colBase = (chunk0 + cc) * 128;
    if (colBase >= Ncols) break;

    for (int idx = tid; idx < 128 * CINP; idx += 256) {
      int cl = idx / CINP, k = idx % CINP;
      long long col = colBase + cl;
      float v = 0.f;
      if (col < Ncols && k < CIN) {
        v = X[(size_t)col * CIN + k];
        if (actMode == 1)      { v = v * sc[k] + sh[k]; v = v > 0.f ? v : 0.f; }
        else if (actMode == 2) { v = v > 0.f ? v : 0.f; }
      }
      Xs[idx] = (half_t)v;
    }
    __syncthreads();

    // prefetch next chunk's activations into cache while WMMAs run
    {
      long long nb = colBase + 128 + tid;
      if (cc + 1 < chunks && nb < Ncols)
        __builtin_prefetch(&X[(size_t)nb * CIN], 0, 1);
    }

    v8f acc[NT] = {};
#pragma unroll
    for (int ks = 0; ks < KSTEPS; ++ks) {
      v16h bx;                                   // B: activations (K x 16 cols)
#pragma unroll
      for (int h = 0; h < 16; ++h)
        bx[h] = Xs[colLocal * CINP + ks * 32 + kmapB(lane, h)];
#pragma unroll
      for (int mt = 0; mt < NT; ++mt) {
        v16h aw;                                 // A: weights (16 outch x K)
        const int mrow = mt * 16 + nlane;
#pragma unroll
        for (int h = 0; h < 16; ++h)
          aw[h] = Ws[mrow * CINP + ks * 32 + kmapA(lane, h)];
        acc[mt] = __builtin_amdgcn_wmma_f32_16x16x32_f16(
            false, aw, false, bx, (short)0, acc[mt], false, false);
      }
    }

    // D: lane -> data column (n), VGPR r -> out channel (mt*16 + off + r):
    // 8 consecutive channels per lane -> two float4 stores per tile.
    const long long col = colBase + colLocal;
    if (col < Ncols) {
      const int off = (lane & 16) ? 8 : 0;
#pragma unroll
      for (int mt = 0; mt < NT; ++mt) {
        float* dst = Y + (size_t)col * COUT + mt * 16 + off;
        float4 lo, hi;
        lo.x = acc[mt][0]; lo.y = acc[mt][1]; lo.z = acc[mt][2]; lo.w = acc[mt][3];
        hi.x = acc[mt][4]; hi.y = acc[mt][5]; hi.z = acc[mt][6]; hi.w = acc[mt][7];
        if (bias) {
          const float* bp = bias + mt * 16 + off;
          lo.x += bp[0]; lo.y += bp[1]; lo.z += bp[2]; lo.w += bp[3];
          hi.x += bp[4]; hi.y += bp[5]; hi.z += bp[6]; hi.w += bp[7];
        }
        *(float4*)(dst)     = lo;
        *(float4*)(dst + 4) = hi;
      }
    }
    __syncthreads();
  }
}

// ===========================================================================
// BatchNorm statistics. X layout [col][C] read LINEARLY (coalesced): C divides
// 256 so each thread's channel (tid % C) is invariant; per-thread register
// accumulation, LDS tree reduction across groups, one global atomic per ch.
// Slot layout: [0:128]=sum [128:256]=sumsq [256:384]=scale [384:512]=shift
// ===========================================================================
__global__ __launch_bounds__(256)
void zero_stats_kernel(float* stats) {
  int i = blockIdx.x * 256 + threadIdx.x;
  if (i < 16 * 512) stats[i] = 0.f;
}

__global__ __launch_bounds__(256)
void stats_partial_kernel(const float* __restrict__ X, int C, long long N,
                          float* __restrict__ stats, int slot) {
  const int tid = threadIdx.x;
  const int c = tid % C;                   // invariant: 256 % C == 0
  const long long total = N * (long long)C;
  long long i = (long long)blockIdx.x * 256 + tid;
  const long long stride = (long long)gridDim.x * 256;
  float s = 0.f, ss = 0.f;
  for (; i < total; i += stride) {
    float v = X[i];
    s += v; ss += v * v;
  }
  __shared__ float rs[256], rss[256];
  rs[tid] = s; rss[tid] = ss;
  __syncthreads();
  for (int o = 128; o >= C; o >>= 1) {     // o % C == 0 throughout
    if (tid < o) { rs[tid] += rs[tid + o]; rss[tid] += rss[tid + o]; }
    __syncthreads();
  }
  if (tid < C) {
    atomicAdd(&stats[(size_t)slot * 512 + c], rs[tid]);
    atomicAdd(&stats[(size_t)slot * 512 + 128 + c], rss[tid]);
  }
}

__global__ __launch_bounds__(128)
void stats_finalize_kernel(float* stats, int slot, int C, float invN,
                           const float* __restrict__ gamma,
                           const float* __restrict__ beta) {
  int c = threadIdx.x;
  if (c < C) {
    float mean = stats[(size_t)slot * 512 + c] * invN;
    float var  = stats[(size_t)slot * 512 + 128 + c] * invN - mean * mean;
    float scl  = gamma[c] * rsqrtf(var + 1e-5f);
    stats[(size_t)slot * 512 + 256 + c] = scl;
    stats[(size_t)slot * 512 + 384 + c] = beta[c] - mean * scl;
  }
}

// ===========================================================================
// bn_fts over channel-major input features (B,16,P) + transpose to [NC][16]
// ===========================================================================
__global__ __launch_bounds__(256)
void fts_stats_kernel(const float* __restrict__ f, const float* __restrict__ mask,
                      const float* __restrict__ g, const float* __restrict__ b,
                      float* __restrict__ stats) {
  const int c = blockIdx.x;  // 16 channels
  float s = 0.f, ss = 0.f;
  for (int i = threadIdx.x; i < NC; i += 256) {
    int bb = i >> 7, p = i & 127;
    float v = f[((size_t)bb * 16 + c) * 128 + p] * mask[i];
    s += v; ss += v * v;
  }
  __shared__ float rs[256], rss[256];
  rs[threadIdx.x] = s; rss[threadIdx.x] = ss;
  __syncthreads();
  for (int o = 128; o > 0; o >>= 1) {
    if (threadIdx.x < o) { rs[threadIdx.x] += rs[threadIdx.x + o];
                           rss[threadIdx.x] += rss[threadIdx.x + o]; }
    __syncthreads();
  }
  if (threadIdx.x == 0) {
    float invN = 1.f / (float)NC;
    float mean = rs[0] * invN, var = rss[0] * invN - mean * mean;
    float scl = g[c] * rsqrtf(var + 1e-5f);
    stats[256 + c] = scl;                 // slot 0
    stats[384 + c] = b[c] - mean * scl;
  }
}

__global__ __launch_bounds__(256)
void fts_norm_kernel(const float* __restrict__ f, const float* __restrict__ mask,
                     const float* __restrict__ stats, float* __restrict__ out) {
  int i = blockIdx.x * 256 + threadIdx.x;
  if (i < NC * 16) {
    int bp = i / 16, c = i % 16;
    int bb = bp >> 7, p = bp & 127;
    float mk = mask[bp];
    float v = f[((size_t)bb * 16 + c) * 128 + p] * mk;
    out[i] = (v * stats[256 + c] + stats[384 + c]) * mk;
  }
}

// ===========================================================================
// Pairwise Lorentz features over tril pairs -> pw[b*NPAIR+t][2]
// ===========================================================================
__global__ __launch_bounds__(256)
void pairwise_kernel(const float* __restrict__ vec, const float* __restrict__ mask,
                     float* __restrict__ pw) {
  long long t = (long long)blockIdx.x * 256 + threadIdx.x;
  if (t >= NPB) return;
  int b = (int)(t / NPAIR), pr = (int)(t % NPAIR);
  int i = (int)((sqrtf(8.f * (float)pr + 1.f) - 1.f) * 0.5f);
  while ((i + 1) * (i + 2) / 2 <= pr) ++i;
  while (i * (i + 1) / 2 > pr) --i;
  int j = pr - i * (i + 1) / 2;
  float mi = mask[b * 128 + i], mj = mask[b * 128 + j];
  const size_t vb = (size_t)b * 4 * 128;
  float ax = vec[vb + 0*128 + i]*mi, ay = vec[vb + 1*128 + i]*mi;
  float az = vec[vb + 2*128 + i]*mi, ae = vec[vb + 3*128 + i]*mi;
  float bx = vec[vb + 0*128 + j]*mj, by = vec[vb + 1*128 + j]*mj;
  float bz = vec[vb + 2*128 + j]*mj, be = vec[vb + 3*128 + j]*mj;
  float ri = 0.5f * log1pf(2.f * az / fmaxf(ae - az, 1e-20f));
  float rj = 0.5f * log1pf(2.f * bz / fmaxf(be - bz, 1e-20f));
  float pi_ = atan2f(ay, ax), pj_ = atan2f(by, bx);
  const float PI = 3.14159265358979f;
  float x = pi_ - pj_ + PI;
  float m = fmodf(x, 2.f * PI); if (m < 0.f) m += 2.f * PI;
  float dphi = m - PI, dr = ri - rj;
  float delta = sqrtf(dr * dr + dphi * dphi);
  float lnd = logf(fmaxf(delta, 1e-8f));
  float sx = ax + bx, sy = ay + by, sz = az + bz, se = ae + be;
  float m2 = se * se - (sx * sx + sy * sy + sz * sz);
  float lnm2 = logf(fmaxf(m2, 1e-8f));
  pw[(size_t)t * 2 + 0] = lnd;
  pw[(size_t)t * 2 + 1] = lnm2;
}

// pair-embed first layer (2 -> 32), bn0 folded via stats slot 1
__global__ __launch_bounds__(256)
void embed1_kernel(const float* __restrict__ pw, const float* __restrict__ w1,
                   const float* __restrict__ b1, const float* __restrict__ stats,
                   float* __restrict__ h1) {
  long long t = (long long)blockIdx.x * 256 + threadIdx.x;
  if (t >= NPB) return;
  const float* sc = stats + 1 * 512 + 256;
  const float* sh = stats + 1 * 512 + 384;
  float x0 = pw[(size_t)t * 2 + 0] * sc[0] + sh[0];
  float x1 = pw[(size_t)t * 2 + 1] * sc[1] + sh[1];
#pragma unroll
  for (int c = 0; c < 32; ++c)
    h1[(size_t)t * 32 + c] = w1[c * 2] * x0 + w1[c * 2 + 1] * x1 + b1[c];
}

// ===========================================================================
// Edge feature builder: E[col][2*CIN] = [xi ; (xn + pair_w*h + pair_b) - xi]
// H2 raw gets bn2 affine + ReLU (slot 3) on the fly. One workgroup per point.
// ===========================================================================
template <int CIN>
__global__ __launch_bounds__(256)
void edge_build_kernel(const float* __restrict__ fts, const float* __restrict__ H2,
                       const float* __restrict__ pairW, const float* __restrict__ pairB,
                       const float* __restrict__ stats, float* __restrict__ E) {
  constexpr int G = 256 / CIN;
  __shared__ float Hs[G * 64];
  __shared__ float xiS[CIN];
  const int bp = blockIdx.x;
  const int b = bp >> 7, p = bp & 127;
  const int tid = threadIdx.x;
  const float* sc = stats + 3 * 512 + 256;
  const float* sh = stats + 3 * 512 + 384;
  if (tid < CIN) xiS[tid] = fts[(size_t)bp * CIN + tid];
  __syncthreads();
  for (int nc = 0; nc < 127; nc += G) {
    for (int idx = tid; idx < G * 64; idx += 256) {
      int g = idx >> 6, k = idx & 63;
      int n = nc + g;
      float v = 0.f;
      if (n < 127) {
        int q = n + (n >= p ? 1 : 0);
        int hi = p > q ? p : q, lo = p > q ? q : p;
        size_t pi = (size_t)hi * (hi + 1) / 2 + lo;
        float raw = H2[((size_t)b * NPAIR + pi) * 64 + k];
        v = raw * sc[k] + sh[k];
        v = v > 0.f ? v : 0.f;
      }
      Hs[idx] = v;
    }
    __syncthreads();
    int g = tid / CIN, c = tid % CIN;
    int n = nc + g;
    if (n < 127) {
      int q = n + (n >= p ? 1 : 0);
      float pc = pairB[c];
#pragma unroll 8
      for (int k = 0; k < 64; ++k) pc += pairW[c * 64 + k] * Hs[g * 64 + k];
      float xic = xiS[c];
      float xnc = fts[((size_t)(b * 128 + q)) * CIN + c] + pc;
      size_t col = (size_t)bp * 127 + n;
      E[col * (2 * CIN) + c]       = xic;
      E[col * (2 * CIN) + CIN + c] = xnc - xic;
    }
    __syncthreads();
  }
}

// attention scores: s = att_w2 . relu(bn(a)) + att_b2
__global__ __launch_bounds__(256)
void att_score_kernel(const float* __restrict__ A, const float* __restrict__ w2,
                      const float* __restrict__ b2, const float* __restrict__ stats,
                      int slot, float* __restrict__ scores, long long N) {
  long long col = (long long)blockIdx.x * 256 + threadIdx.x;
  if (col >= N) return;
  const float* sc = stats + (size_t)slot * 512 + 256;
  const float* sh = stats + (size_t)slot * 512 + 384;
  float s = b2[0];
#pragma unroll
  for (int k = 0; k < 32; ++k) {
    float v = A[(size_t)col * 32 + k] * sc[k] + sh[k];
    s += w2[k] * (v > 0.f ? v : 0.f);
  }
  scores[col] = s;
}

// softmax over 127 neighbors, attention pooling of relu(bn(x3)), + bn'd
// shortcut, ReLU, mask.  One workgroup (128 thr) per point.
template <int C>
__global__ __launch_bounds__(128)
void softmax_pool_kernel(const float* __restrict__ X3, int slotX,
                         const float* __restrict__ scores,
                         const float* __restrict__ SC, int slotSC,
                         const float* __restrict__ stats,
                         const float* __restrict__ mask, float* __restrict__ out) {
  const int bp = blockIdx.x;
  const int tid = threadIdx.x;
  __shared__ float wts[128];
  __shared__ float red[128];
  const size_t base = (size_t)bp * 127;
  float s = (tid < 127) ? scores[base + tid] : -3.4e38f;
  red[tid] = s; __syncthreads();
  for (int o = 64; o > 0; o >>= 1) {
    if (tid < o) red[tid] = fmaxf(red[tid], red[tid + o]);
    __syncthreads();
  }
  float mx = red[0]; __syncthreads();
  float e = (tid < 127) ? __expf(s - mx) : 0.f;
  red[tid] = e; __syncthreads();
  for (int o = 64; o > 0; o >>= 1) {
    if (tid < o) red[tid] += red[tid + o];
    __syncthreads();
  }
  wts[tid] = e / red[0];
  __syncthreads();
  if (tid < C) {
    const float* scx = stats + (size_t)slotX * 512 + 256;
    const float* shx = stats + (size_t)slotX * 512 + 384;
    float pooled = 0.f;
    for (int n = 0; n < 127; ++n) {
      float v = X3[(base + n) * C + tid] * scx[tid] + shx[tid];
      pooled += wts[n] * (v > 0.f ? v : 0.f);
    }
    float scv = SC[(size_t)bp * C + tid] * stats[(size_t)slotSC * 512 + 256 + tid]
              + stats[(size_t)slotSC * 512 + 384 + tid];
    out[(size_t)bp * C + tid] = fmaxf(scv + pooled, 0.f) * mask[bp];
  }
}

__global__ __launch_bounds__(256)
void concat_kernel(const float* __restrict__ f1, const float* __restrict__ f2,
                   float* __restrict__ out) {
  int i = blockIdx.x * 256 + threadIdx.x;
  if (i < NC * 96) {
    int bp = i / 96, c = i % 96;
    out[i] = (c < 32) ? f1[(size_t)bp * 32 + c] : f2[(size_t)bp * 64 + (c - 32)];
  }
}

// masked mean over points of relu(bn(fus))
__global__ __launch_bounds__(128)
void fuse_mean_kernel(const float* __restrict__ F, const float* __restrict__ stats,
                      const float* __restrict__ mask, float* __restrict__ out) {
  const int b = blockIdx.x, c = threadIdx.x;  // 64 x 128
  const float* sc = stats + 14 * 512 + 256;
  const float* sh = stats + 14 * 512 + 384;
  float cnt = 0.f, s = 0.f;
  for (int p = 0; p < 128; ++p) {
    float mk = mask[b * 128 + p];
    cnt += mk;
    float v = F[((size_t)b * 128 + p) * 128 + c] * sc[c] + sh[c];
    s += fmaxf(v, 0.f) * mk;
  }
  out[(size_t)b * 128 + c] = s / fmaxf(cnt, 1.f);
}

// ===========================================================================
// Host orchestration
// ===========================================================================
extern "C" void kernel_launch(void* const* d_in, const int* in_sizes, int n_in,
                              void* d_out, int out_size, void* d_ws, size_t ws_size,
                              hipStream_t stream) {
  (void)in_sizes; (void)n_in; (void)out_size; (void)ws_size;
  auto IN = [&](int i) { return (const float*)d_in[i]; };
  float* ws = (float*)d_ws;
  float* STATS  = ws + OFF_STATS;
  float* FTS0   = ws + OFF_FTS0;
  float* FTS1   = ws + OFF_FTS1;
  float* FTS2   = ws + OFF_FTS2;
  float* FUSIN  = ws + OFF_FUSIN;
  float* FUSRAW = ws + OFF_FUSRAW;
  float* POOLB  = ws + OFF_POOLB;
  float* FC1O   = ws + OFF_FC1;
  float* SCORES = ws + OFF_SCORES;
  float* SCRAW  = ws + OFF_SCRAW;
  float* PW     = ws + OFF_PW;
  float* H1     = ws + OFF_H1;
  float* H2     = ws + OFF_H2;
  float* BUFA   = ws + OFF_BUFA;
  float* BUFB   = ws + OFF_BUFB;

  // param leaf indices (jax pytree sorted-key flatten order; 0..2 = inputs)
  const float* features = IN(0);
  const float* vectors  = IN(1);
  const float* mask     = IN(2);

  auto stats = [&](const float* X, int C, long long N, int slot, int gi, int bi) {
    stats_partial_kernel<<<256, 256, 0, stream>>>(X, C, N, STATS, slot);
    stats_finalize_kernel<<<1, 128, 0, stream>>>(STATS, slot, C,
        (float)(1.0 / (double)N), IN(gi), IN(bi));
  };

  // shared-mem sizes: (W + X tile) halves
  auto shmem = [](int cout, int cinp) { return (size_t)(cout + 128) * cinp * 2; };

  zero_stats_kernel<<<32, 256, 0, stream>>>(STATS);

  // ---- bn_fts + transpose ----
  fts_stats_kernel<<<16, 256, 0, stream>>>(features, mask, IN(41), IN(42), STATS);
  fts_norm_kernel<<<(NC * 16 + 255) / 256, 256, 0, stream>>>(features, mask, STATS, FTS0);

  // ---- pairwise features + pair embed ----
  pairwise_kernel<<<(unsigned)((NPB + 255) / 256), 256, 0, stream>>>(vectors, mask, PW);
  stats(PW, 2, NPB, 1, 50, 51);
  embed1_kernel<<<(unsigned)((NPB + 255) / 256), 256, 0, stream>>>(PW, IN(56), IN(48), STATS, H1);
  stats(H1, 32, NPB, 2, 52, 53);
  gemm_wmma_kernel<32, 32, 64><<<1032, 256, shmem(64, 32), stream>>>(
      H1, H2, IN(57), IN(49), STATS, 2, 1, NPB, 4);
  stats(H2, 64, NPB, 3, 54, 55);

  const unsigned GNE = 2032;  // NE/128/4
  // ================= EdgeConv block 0 (16 -> 32) =================
  edge_build_kernel<16><<<NC, 256, 0, stream>>>(FTS0, H2, IN(18), IN(17), STATS, BUFA);
  gemm_wmma_kernel<32, 32, 32><<<GNE, 256, shmem(32, 32), stream>>>(
      BUFA, BUFB, IN(14), nullptr, STATS, -1, 0, NE, 4);
  stats(BUFB, 32, NE, 4, 8, 9);
  gemm_wmma_kernel<32, 32, 32><<<GNE, 256, shmem(32, 32), stream>>>(
      BUFB, BUFA, IN(15), nullptr, STATS, 4, 1, NE, 4);
  stats(BUFA, 32, NE, 5, 10, 11);
  gemm_wmma_kernel<32, 32, 32><<<GNE, 256, shmem(32, 32), stream>>>(
      BUFA, BUFB, IN(16), nullptr, STATS, 5, 1, NE, 4);
  stats(BUFB, 32, NE, 6, 12, 13);
  // attention
  gemm_wmma_kernel<32, 32, 32><<<GNE, 256, shmem(32, 32), stream>>>(
      BUFB, BUFA, IN(6), nullptr, STATS, 6, 1, NE, 4);
  stats(BUFA, 32, NE, 7, 4, 5);
  att_score_kernel<<<(unsigned)((NE + 255) / 256), 256, 0, stream>>>(
      BUFA, IN(7), IN(3), STATS, 7, SCORES, NE);
  // shortcut
  gemm_wmma_kernel<16, 32, 32><<<64, 256, shmem(32, 32), stream>>>(
      FTS0, SCRAW, IN(21), nullptr, STATS, -1, 0, NC, 1);
  stats(SCRAW, 32, NC, 8, 19, 20);
  softmax_pool_kernel<32><<<NC, 128, 0, stream>>>(BUFB, 6, SCORES, SCRAW, 8, STATS, mask, FTS1);

  // ================= EdgeConv block 1 (32 -> 64) =================
  edge_build_kernel<32><<<NC, 256, 0, stream>>>(FTS1, H2, IN(37), IN(36), STATS, BUFA);
  gemm_wmma_kernel<64, 64, 64><<<GNE, 256, shmem(64, 64), stream>>>(
      BUFA, BUFB, IN(33), nullptr, STATS, -1, 0, NE, 4);
  stats(BUFB, 64, NE, 9, 27, 28);
  gemm_wmma_kernel<64, 64, 64><<<GNE, 256, shmem(64, 64), stream>>>(
      BUFB, BUFA, IN(34), nullptr, STATS, 9, 1, NE, 4);
  stats(BUFA, 64, NE, 10, 29, 30);
  gemm_wmma_kernel<64, 64, 64><<<GNE, 256, shmem(64, 64), stream>>>(
      BUFA, BUFB, IN(35), nullptr, STATS, 10, 1, NE, 4);
  stats(BUFB, 64, NE, 11, 31, 32);
  gemm_wmma_kernel<64, 64, 32><<<GNE, 256, shmem(32, 64), stream>>>(
      BUFB, BUFA, IN(25), nullptr, STATS, 11, 1, NE, 4);
  stats(BUFA, 32, NE, 12, 23, 24);
  att_score_kernel<<<(unsigned)((NE + 255) / 256), 256, 0, stream>>>(
      BUFA, IN(26), IN(22), STATS, 12, SCORES, NE);
  gemm_wmma_kernel<32, 32, 64><<<64, 256, shmem(64, 32), stream>>>(
      FTS1, SCRAW, IN(40), nullptr, STATS, -1, 0, NC, 1);
  stats(SCRAW, 64, NC, 13, 38, 39);
  softmax_pool_kernel<64><<<NC, 128, 0, stream>>>(BUFB, 11, SCORES, SCRAW, 13, STATS, mask, FTS2);

  // ================= fusion + head =================
  concat_kernel<<<(NC * 96 + 255) / 256, 256, 0, stream>>>(FTS1, FTS2, FUSIN);
  gemm_wmma_kernel<96, 96, 128><<<64, 256, shmem(128, 96), stream>>>(
      FUSIN, FUSRAW, IN(47), nullptr, STATS, -1, 0, NC, 1);
  stats(FUSRAW, 128, NC, 14, 45, 46);
  fuse_mean_kernel<<<B_, 128, 0, stream>>>(FUSRAW, STATS, mask, POOLB);
  gemm_wmma_kernel<128, 128, 256><<<1, 256, shmem(256, 128), stream>>>(
      POOLB, FC1O, IN(43), nullptr, STATS, -1, 0, (long long)B_, 1);
  gemm_wmma_kernel<256, 256, 128><<<1, 256, shmem(128, 256), stream>>>(
      FC1O, (float*)d_out, IN(44), nullptr, STATS, -1, 2, (long long)B_, 1);
}
